// BigBirdAttention_36687610643078
// MI455X (gfx1250) — matrix operand order repaired and checked by reading, not compiled, not run on hardware
//
#include <hip/hip_runtime.h>

// Problem constants (match reference)
constexpr int kDim   = 1024;
constexpr int kHeads = 16;
constexpr int kHDim  = 64;
constexpr int kBlk   = 64;
constexpr int kSeq   = 2048;
constexpr int kBatch = 2;
constexpr int kNB    = kSeq / kBlk;      // 32 blocks
constexpr int kMTok  = kBatch * kSeq;    // 4096 token rows

typedef _Float16 v16h __attribute__((ext_vector_type(16)));
typedef _Float16 v8h  __attribute__((ext_vector_type(8)));
typedef float    v8f  __attribute__((ext_vector_type(8)));

static __device__ inline v16h cat8(v8h a, v8h b) {
  return __builtin_shufflevector(a, b, 0,1,2,3,4,5,6,7,8,9,10,11,12,13,14,15);
}
static __device__ inline v8f wmma_f16(v16h a, v16h b, v8f c) {
  // D = A(16x32 f16) * B(32x16 f16) + C(16x16 f32)
  return __builtin_amdgcn_wmma_f32_16x16x32_f16(false, a, false, b, (short)0, c, false, false);
}
// reductions across the 16 lanes that share the same C-matrix rows (wave32 halves)
static __device__ inline float rmax16(float v) {
  v = fmaxf(v, __shfl_xor(v, 1, 32));
  v = fmaxf(v, __shfl_xor(v, 2, 32));
  v = fmaxf(v, __shfl_xor(v, 4, 32));
  v = fmaxf(v, __shfl_xor(v, 8, 32));
  return v;
}
static __device__ inline float rsum16(float v) {
  v += __shfl_xor(v, 1, 32);
  v += __shfl_xor(v, 2, 32);
  v += __shfl_xor(v, 4, 32);
  v += __shfl_xor(v, 8, 32);
  return v;
}

__global__ void cvt_f16_kernel(const float* __restrict__ in, _Float16* __restrict__ out, int n) {
  int i = (blockIdx.x * blockDim.x + threadIdx.x) * 8;
  const int stride = gridDim.x * blockDim.x * 8;
  for (; i < n; i += stride) {
    const float4 f0 = *(const float4*)(in + i);
    const float4 f1 = *(const float4*)(in + i + 4);
    v8h h;
    h[0] = (_Float16)f0.x; h[1] = (_Float16)f0.y; h[2] = (_Float16)f0.z; h[3] = (_Float16)f0.w;
    h[4] = (_Float16)f1.x; h[5] = (_Float16)f1.y; h[6] = (_Float16)f1.z; h[7] = (_Float16)f1.w;
    *(v8h*)(out + i) = h;
  }
}

// out = A(f16, M x K rowmajor) @ W(f16, N x K rowmajor)^T + bias
// Workgroup = 4 waves; macro-tile 256(M) x 64(N); each wave owns a 64x64 block
// (4x4 grid of 16x16 WMMA accumulators -> 16 WMMAs per 8 fragment fetches).
// W tile (64 x 32 per K-step) is staged into LDS with global_load_async_to_lds_b128,
// double-buffered, pipelined on ASYNCcnt (CDNA5 async-to-LDS path).
// MODE 0: f16 row-major out | MODE 1: f16 head-transposed vT[b][h][d][s] | MODE 2: f32 out
template <int MODE>
__global__ __launch_bounds__(128)
void gemm_xwT_kernel(const _Float16* __restrict__ A,
                     const _Float16* __restrict__ W,
                     const float* __restrict__ bias,
                     void* __restrict__ out, int K) {
  constexpr int LDB = 40;  // halfs per LDS row: 80B (16B-aligned, 20-bank stride)
  __shared__ __align__(16) _Float16 bbuf[2][64 * LDB];

  const int t     = threadIdx.x;      // 0..127
  const int lane  = t & 31;
  const int wv    = t >> 5;
  const int half_ = lane >> 4;        // 0 or 1
  const int l16   = lane & 15;
  const int nBase = blockIdx.x * 64;
  const int mBase = blockIdx.y * 256 + wv * 64;

  // async staging: 64 rows x 32 halfs = 4KB per K-step; 128 threads x 2 x 16B chunks
  const int c0 = t * 2, c1 = t * 2 + 1;            // chunk ids 0..255
  const int sn0 = c0 >> 2, sk0 = (c0 & 3) * 8;     // row, half-offset within 32
  const int sn1 = c1 >> 2, sk1 = (c1 & 3) * 8;
  const _Float16* wbase = W + (size_t)nBase * K;

  auto issue = [&](int buf, int k0) {
    const unsigned l0 = (unsigned)(unsigned long long)&bbuf[buf][sn0 * LDB + sk0];
    const unsigned l1 = (unsigned)(unsigned long long)&bbuf[buf][sn1 * LDB + sk1];
    const unsigned long long g0 = (unsigned long long)(wbase + (size_t)sn0 * K + k0 + sk0);
    const unsigned long long g1 = (unsigned long long)(wbase + (size_t)sn1 * K + k0 + sk1);
    asm volatile("global_load_async_to_lds_b128 %0, %1, off\n\t"
                 "global_load_async_to_lds_b128 %2, %3, off"
                 :: "v"(l0), "v"(g0), "v"(l1), "v"(g1) : "memory");
  };

  // A row pointers: A-frag rows use M = lane%16 within each 16-row tile
  const _Float16* arow[4];
#pragma unroll
  for (int mt = 0; mt < 4; ++mt)
    arow[mt] = A + (size_t)(mBase + mt * 16 + l16) * K;

  v8f acc[4][4];
#pragma unroll
  for (int mt = 0; mt < 4; ++mt)
#pragma unroll
    for (int nt = 0; nt < 4; ++nt)
      acc[mt][nt] = (v8f){0.f, 0.f, 0.f, 0.f, 0.f, 0.f, 0.f, 0.f};

  issue(0, 0);  // prologue fill
  const int kSteps = K / 32;
  for (int kk = 0; kk < kSteps; ++kk) {
    asm volatile("s_wait_asynccnt 0x0" ::: "memory");  // own fills landed
    __syncthreads();                                   // all waves' fills landed / reads done
    if (kk + 1 < kSteps) issue((kk + 1) & 1, (kk + 1) * 32);

    const _Float16* bb = &bbuf[kk & 1][0];
    v16h bfrag[4];
#pragma unroll
    for (int nt = 0; nt < 4; ++nt) {
      const _Float16* br = bb + (nt * 16 + l16) * LDB + half_ * 8;
      bfrag[nt] = cat8(*(const v8h*)br, *(const v8h*)(br + 16));
    }
    const int k0 = kk * 32;
    v16h afrag[4];
#pragma unroll
    for (int mt = 0; mt < 4; ++mt) {
      const _Float16* ar = arow[mt] + k0 + half_ * 8;
      afrag[mt] = cat8(*(const v8h*)ar, *(const v8h*)(ar + 16));
    }
#pragma unroll
    for (int mt = 0; mt < 4; ++mt)
#pragma unroll
      for (int nt = 0; nt < 4; ++nt)
        acc[mt][nt] = wmma_f16(afrag[mt], bfrag[nt], acc[mt][nt]);
  }

#pragma unroll
  for (int nt = 0; nt < 4; ++nt) {
    const int nGlob = nBase + nt * 16 + l16;
    const float bn = bias[nGlob];
#pragma unroll
    for (int mt = 0; mt < 4; ++mt) {
#pragma unroll
      for (int i = 0; i < 8; ++i) {
        const int mGlob = mBase + mt * 16 + half_ * 8 + i;  // C rows: M = half*8 + vgpr
        const float v = acc[mt][nt][i] + bn;
        if (MODE == 0) {
          ((_Float16*)out)[(size_t)mGlob * kDim + nGlob] = (_Float16)v;
        } else if (MODE == 2) {
          ((float*)out)[(size_t)mGlob * kDim + nGlob] = v;
        } else {
          // replicate reference's .view head split: (b, h, s, d) from (row, col)
          const int b = mGlob >> 11, r = mGlob & (kSeq - 1);
          const int h = r >> 7, rr = r & 127;
          const int chunk = nGlob >> 6, d = nGlob & 63;
          const int s = rr * 16 + chunk;
          ((_Float16*)out)[(((size_t)(b * kHeads + h) * kHDim + d) << 11) + s] = (_Float16)v;
        }
      }
    }
  }
}

__global__ __launch_bounds__(128)
void bigbird_attn_kernel(const _Float16* __restrict__ qh,
                         const _Float16* __restrict__ kh,
                         const _Float16* __restrict__ vT,
                         const int* __restrict__ srcB,
                         const int* __restrict__ tgtB,
                         _Float16* __restrict__ attnOut) {
  __shared__ __align__(16) _Float16 pLds[4][16][80];  // per-wave 16x64 P tile
  const int lane  = threadIdx.x & 31;
  const int wv    = threadIdx.x >> 5;
  const int half_ = lane >> 4;
  const int l16   = lane & 15;
  const int qb = blockIdx.x;  // query block
  const int h  = blockIdx.y;
  const int b  = blockIdx.z;

  const int s0a = srcB[0], s1a = srcB[1], s2a = srcB[2];
  const int t0a = tgtB[0], t1a = tgtB[1], t2a = tgtB[2];

  // Q fragments for this wave's 16 query rows (head-dim K = 64 -> two A frags)
  const int qs = qb * kBlk + wv * 16 + l16;
  const _Float16* qrow =
      qh + ((size_t)(b * kSeq) + h * 128 + (qs >> 4)) * (size_t)kDim + (qs & 15) * kHDim;
  const v16h a0 = cat8(*(const v8h*)(qrow + half_ * 8),      *(const v8h*)(qrow + half_ * 8 + 16));
  const v16h a1 = cat8(*(const v8h*)(qrow + 32 + half_ * 8), *(const v8h*)(qrow + 32 + half_ * 8 + 16));

  float M[8], L[8];
  v8f O[4];
#pragma unroll
  for (int i = 0; i < 8; ++i) { M[i] = -3.0e38f; L[i] = 0.f; }
#pragma unroll
  for (int dt = 0; dt < 4; ++dt) O[dt] = (v8f){0.f, 0.f, 0.f, 0.f, 0.f, 0.f, 0.f, 0.f};

  for (int kb = 0; kb < kNB; ++kb) {
    int dd = qb - kb; if (dd < 0) dd = -dd;
    const bool allowed = (dd <= 1) || (kb == 0) || (kb == kNB - 1) ||
                         (s0a == qb && t0a == kb) ||
                         (s1a == qb && t1a == kb) ||
                         (s2a == qb && t2a == kb);
    if (!allowed) continue;  // uniform branch: EXEC stays all-ones for WMMA

    // ---- scores: 4 tiles of 16x16, K = head dim 64 ----
    float sc[4][8];
#pragma unroll
    for (int nt = 0; nt < 4; ++nt) {
      const int ks = kb * kBlk + nt * 16 + l16;
      const _Float16* krow =
          kh + ((size_t)(b * kSeq) + h * 128 + (ks >> 4)) * (size_t)kDim + (ks & 15) * kHDim;
      const v16h b0 = cat8(*(const v8h*)(krow + half_ * 8),      *(const v8h*)(krow + half_ * 8 + 16));
      const v16h b1 = cat8(*(const v8h*)(krow + 32 + half_ * 8), *(const v8h*)(krow + 32 + half_ * 8 + 16));
      v8f cc = {0.f, 0.f, 0.f, 0.f, 0.f, 0.f, 0.f, 0.f};
      cc = wmma_f16(a0, b0, cc);
      cc = wmma_f16(a1, b1, cc);
#pragma unroll
      for (int i = 0; i < 8; ++i) sc[nt][i] = cc[i] * 0.125f;  // 1/sqrt(64)
    }

    // ---- online softmax update ----
#pragma unroll
    for (int i = 0; i < 8; ++i) {
      float pm = fmaxf(fmaxf(sc[0][i], sc[1][i]), fmaxf(sc[2][i], sc[3][i]));
      pm = rmax16(pm);
      const float nm = fmaxf(M[i], pm);
      const float alpha = __expf(M[i] - nm);
      M[i] = nm;
      L[i] *= alpha;
#pragma unroll
      for (int dt = 0; dt < 4; ++dt) O[dt][i] *= alpha;
    }

    float ps[8];
#pragma unroll
    for (int i = 0; i < 8; ++i) ps[i] = 0.f;
#pragma unroll
    for (int nt = 0; nt < 4; ++nt) {
#pragma unroll
      for (int i = 0; i < 8; ++i) {
        const float p = __expf(sc[nt][i] - M[i]);
        ps[i] += p;
        pLds[wv][half_ * 8 + i][nt * 16 + l16] = (_Float16)p;  // C-layout -> row-major in LDS
      }
    }
    asm volatile("s_wait_dscnt 0x0" ::: "memory");  // CDNA5 split DS counter
#pragma unroll
    for (int i = 0; i < 8; ++i) L[i] += rsum16(ps[i]);

    // ---- P @ V : A frags from LDS (transposed), B frags from vT (contig in s) ----
    const _Float16* prow = &pLds[wv][l16][0];
    const v16h pa0 = cat8(*(const v8h*)(prow + half_ * 8),      *(const v8h*)(prow + half_ * 8 + 16));
    const v16h pa1 = cat8(*(const v8h*)(prow + 32 + half_ * 8), *(const v8h*)(prow + 32 + half_ * 8 + 16));
#pragma unroll
    for (int dt = 0; dt < 4; ++dt) {
      const _Float16* vrow =
          vT + (((size_t)(b * kHeads + h) * kHDim + dt * 16 + l16) << 11) + kb * kBlk;
      const v16h vb0 = cat8(*(const v8h*)(vrow + half_ * 8),      *(const v8h*)(vrow + half_ * 8 + 16));
      const v16h vb1 = cat8(*(const v8h*)(vrow + 32 + half_ * 8), *(const v8h*)(vrow + 32 + half_ * 8 + 16));
      O[dt] = wmma_f16(pa0, vb0, O[dt]);
      O[dt] = wmma_f16(pa1, vb1, O[dt]);
    }
  }

  // ---- normalize and store in transpose(0,2,1,3)-merged layout: col = h*64 + d ----
  float inv[8];
#pragma unroll
  for (int i = 0; i < 8; ++i) inv[i] = 1.0f / L[i];
  const int srow = qb * kBlk + wv * 16;
#pragma unroll
  for (int dt = 0; dt < 4; ++dt) {
#pragma unroll
    for (int i = 0; i < 8; ++i) {
      const int s = srow + half_ * 8 + i;
      const int col = h * kHDim + dt * 16 + l16;
      attnOut[((size_t)(b * kSeq) + s) * kDim + col] = (_Float16)(O[dt][i] * inv[i]);
    }
  }
}

extern "C" void kernel_launch(void* const* d_in, const int* in_sizes, int n_in,
                              void* d_out, int out_size, void* d_ws, size_t ws_size,
                              hipStream_t stream) {
  const float* x  = (const float*)d_in[0];
  const float* Wq = (const float*)d_in[1];
  const float* bq = (const float*)d_in[2];
  const float* Wk = (const float*)d_in[3];
  const float* bk = (const float*)d_in[4];
  const float* Wv = (const float*)d_in[5];
  const float* bv = (const float*)d_in[6];
  const float* Wo = (const float*)d_in[7];
  const float* bo = (const float*)d_in[8];
  const int* srcB = (const int*)d_in[9];
  const int* tgtB = (const int*)d_in[10];

  // workspace partition (f16): 5 activations * 8MB + 4 weights * 2MB = 48 MB
  _Float16* ws = (_Float16*)d_ws;
  const size_t NX = (size_t)kMTok * kDim;  // 4M elements
  const size_t NW = (size_t)kDim * kDim;   // 1M elements
  _Float16* xh   = ws;
  _Float16* qh   = xh + NX;
  _Float16* khp  = qh + NX;
  _Float16* vTh  = khp + NX;
  _Float16* atth = vTh + NX;
  _Float16* Wqh  = atth + NX;
  _Float16* Wkh  = Wqh + NW;
  _Float16* Wvh  = Wkh + NW;
  _Float16* Woh  = Wvh + NW;

  cvt_f16_kernel<<<1024, 256, 0, stream>>>(x, xh, (int)NX);
  cvt_f16_kernel<<<512, 256, 0, stream>>>(Wq, Wqh, (int)NW);
  cvt_f16_kernel<<<512, 256, 0, stream>>>(Wk, Wkh, (int)NW);
  cvt_f16_kernel<<<512, 256, 0, stream>>>(Wv, Wvh, (int)NW);
  cvt_f16_kernel<<<512, 256, 0, stream>>>(Wo, Woh, (int)NW);

  dim3 gblk(128);
  dim3 ggrid(kDim / 64, kMTok / 256);  // (16, 16): 4 waves/block, 64x64 tile per wave
  gemm_xwT_kernel<0><<<ggrid, gblk, 0, stream>>>(xh, Wqh, bq, qh, kDim);
  gemm_xwT_kernel<0><<<ggrid, gblk, 0, stream>>>(xh, Wkh, bk, khp, kDim);
  gemm_xwT_kernel<1><<<ggrid, gblk, 0, stream>>>(xh, Wvh, bv, vTh, kDim);

  bigbird_attn_kernel<<<dim3(kNB, kHeads, kBatch), 128, 0, stream>>>(qh, khp, vTh, srcB, tgtB, atth);

  gemm_xwT_kernel<2><<<ggrid, gblk, 0, stream>>>(atth, Woh, bo, d_out, kDim);
}
